// MoELayer_40716289966661
// MI455X (gfx1250) — matrix operand order repaired and checked by reading, not compiled, not run on hardware
//
#include <hip/hip_runtime.h>

typedef __attribute__((ext_vector_type(16))) __bf16 v16bf;
typedef __attribute__((ext_vector_type(8)))  float  v8f;
typedef __attribute__((ext_vector_type(4)))  int    v4i;

#define TOK    4096
#define HID    1024
#define INTERF 4096
#define NEXP   8
#define NSLOT  8192
#define DUMMY  8192

#if __has_builtin(__builtin_amdgcn_global_load_async_to_lds_b128) && \
    __has_builtin(__builtin_amdgcn_s_wait_asynccnt)
#define HAVE_ASYNC_LDS 1
#endif

__device__ __forceinline__ void ld_async_b128(const unsigned short* g, unsigned short* l) {
#ifdef HAVE_ASYNC_LDS
  __builtin_amdgcn_global_load_async_to_lds_b128((v4i*)g, (v4i*)l, 0, 0);
#else
  (void)g; (void)l;
#endif
}

__device__ __forceinline__ void wait_async0() {
#ifdef HAVE_ASYNC_LDS
  __builtin_amdgcn_s_wait_asynccnt(0);
#endif
}

__device__ __forceinline__ unsigned short f2bf(float f) {
  unsigned int u = __float_as_uint(f);
  u += 0x7FFFu + ((u >> 16) & 1u);           // round-to-nearest-even
  return (unsigned short)(u >> 16);
}

// ---------------------------------------------------------------- converts
__global__ void cvt_bf16_kernel(const float* __restrict__ src,
                                unsigned short* __restrict__ dst, int n) {
  int i = blockIdx.x * blockDim.x + threadIdx.x;
  int stride = gridDim.x * blockDim.x;
  for (; i < n; i += stride) dst[i] = f2bf(src[i]);
}

// ---------------------------------------------------------------- init
__global__ void init_kernel(int* __restrict__ list, int* __restrict__ cnt) {
  int t = threadIdx.x;
  if (t < NEXP) cnt[t] = 0;
  for (int i = t; i < NEXP * TOK; i += blockDim.x) list[i] = DUMMY;
}

// ---------------------------------------------------------------- router
__global__ void router_kernel(const float* __restrict__ x,
                              const float* __restrict__ Wg,
                              int* __restrict__ cnt, int* __restrict__ list,
                              float* __restrict__ slot_w, float* __restrict__ ent) {
  const int lane = threadIdx.x & 31;
  const int token = blockIdx.x * (blockDim.x >> 5) + (threadIdx.x >> 5);
  if (token >= TOK) return;
  float acc[NEXP];
#pragma unroll
  for (int e = 0; e < NEXP; ++e) acc[e] = 0.f;
  const float* xr = x + (size_t)token * HID;
  for (int d = lane; d < HID; d += 32) {
    float xv = xr[d];
    const float* wr = Wg + d * NEXP;
#pragma unroll
    for (int e = 0; e < NEXP; ++e) acc[e] += xv * wr[e];
  }
#pragma unroll
  for (int e = 0; e < NEXP; ++e) {
#pragma unroll
    for (int off = 16; off > 0; off >>= 1) acc[e] += __shfl_xor(acc[e], off, 32);
  }
  if (lane == 0) {
    float m = acc[0];
#pragma unroll
    for (int e = 1; e < NEXP; ++e) m = fmaxf(m, acc[e]);
    float p[NEXP], sum = 0.f;
#pragma unroll
    for (int e = 0; e < NEXP; ++e) { p[e] = __expf(acc[e] - m); sum += p[e]; }
    float inv = 1.f / sum;
    float H = 0.f;
#pragma unroll
    for (int e = 0; e < NEXP; ++e) { p[e] *= inv; H -= p[e] * __logf(p[e] + 1e-8f); }
    ent[token] = H;
    int i0 = 0; float v0 = p[0];
#pragma unroll
    for (int e = 1; e < NEXP; ++e) if (p[e] > v0) { v0 = p[e]; i0 = e; }
    int i1 = -1; float v1 = -1.f;
#pragma unroll
    for (int e = 0; e < NEXP; ++e) if (e != i0 && p[e] > v1) { v1 = p[e]; i1 = e; }
    float wsum = v0 + v1;
    float w0 = v0 / wsum, w1 = v1 / wsum;
    int s0 = token * 2, s1 = token * 2 + 1;
    slot_w[s0] = w0; slot_w[s1] = w1;
    int j0 = atomicAdd(&cnt[i0], 1); list[i0 * TOK + j0] = s0;
    int j1 = atomicAdd(&cnt[i1], 1); list[i1 * TOK + j1] = s1;
  }
}

// ---------------------------------------------------------------- GEMM1: H = swish(Xg @ W1[e])
__global__ __launch_bounds__(256)
void gemm1_kernel(const unsigned short* __restrict__ Xb,
                  const unsigned short* __restrict__ W1b,
                  unsigned short* __restrict__ Hb,
                  const int* __restrict__ list, const int* __restrict__ cnt) {
  const int e = blockIdx.z;
  const int Ne = cnt[e];
  const int mBase = blockIdx.x * 128;
  if (mBase >= Ne) return;
  const int nBase = blockIdx.y * 128;

  __shared__ unsigned short sA[2][128 * 32];
  __shared__ unsigned short sB[2][128 * 32];
  __shared__ int sSlot[128];

  const int t = threadIdx.x;
  if (t < 128) sSlot[t] = list[e * TOK + mBase + t];
  __syncthreads();

  const int ar = t >> 1, ah = (t & 1) * 16;        // A stage: row, 16-col half
  const int bk = t >> 3, bn = (t & 7) * 16;        // B stage: k row, 16-col chunk
  const int aslot = sSlot[ar];
  const int atok = (aslot >= DUMMY) ? 0 : (aslot >> 1);
  const unsigned short* aSrc = Xb + (size_t)atok * HID + ah;
  const unsigned short* bSrc = W1b + (size_t)e * HID * INTERF + (size_t)bk * INTERF + nBase + bn;

  const int lane = t & 31;
  const int wid  = t >> 5;
  const int wm = (wid & 3) * 32;                   // 4 waves along M
  const int wn = (wid >> 2) * 64;                  // 2 waves along N
  const int lrow  = lane & 15;
  const int kselA = (lane < 16) ? 0 : 8;           // ISA 16-bit A layout
  const int kselB = (lane < 16) ? 0 : 8;           // ISA 16-bit B layout

  union Frag { uint4 u[2]; v16bf v; unsigned short s[16]; };
  v8f acc[2][4];
  const v8f vzero = {0.f,0.f,0.f,0.f,0.f,0.f,0.f,0.f};
#pragma unroll
  for (int mi = 0; mi < 2; ++mi)
#pragma unroll
    for (int ni = 0; ni < 4; ++ni) acc[mi][ni] = vzero;

  // ---- prologue: stage step 0 into buffer 0
#ifdef HAVE_ASYNC_LDS
  ld_async_b128(aSrc,     &sA[0][ar * 32 + ah]);
  ld_async_b128(aSrc + 8, &sA[0][ar * 32 + ah + 8]);
#else
  {
    uint4 a0 = ((const uint4*)aSrc)[0], a1 = ((const uint4*)aSrc)[1];
    ((uint4*)&sA[0][ar * 32 + ah])[0] = a0;
    ((uint4*)&sA[0][ar * 32 + ah])[1] = a1;
  }
#endif
  {
    Frag bw;
    bw.u[0] = ((const uint4*)bSrc)[0];
    bw.u[1] = ((const uint4*)bSrc)[1];
#pragma unroll
    for (int i = 0; i < 16; ++i) sB[0][(bn + i) * 32 + bk] = bw.s[i];
  }
  wait_async0();
  __syncthreads();

  const int KSTEPS = HID / 32;
  for (int s = 0; s < KSTEPS; ++s) {
    const int cur = s & 1, nxt = cur ^ 1;
    const bool more = (s + 1) < KSTEPS;

    Frag bw;
    if (more) {                                     // early-issue next-step loads
      const unsigned short* bp = bSrc + (size_t)(s + 1) * 32 * INTERF;
      bw.u[0] = ((const uint4*)bp)[0];
      bw.u[1] = ((const uint4*)bp)[1];
#ifdef HAVE_ASYNC_LDS
      const unsigned short* ap = aSrc + (s + 1) * 32;
      ld_async_b128(ap,     &sA[nxt][ar * 32 + ah]);
      ld_async_b128(ap + 8, &sA[nxt][ar * 32 + ah + 8]);
#endif
    }
#ifndef HAVE_ASYNC_LDS
    uint4 a0n, a1n;
    if (more) {
      const unsigned short* ap = aSrc + (s + 1) * 32;
      a0n = ((const uint4*)ap)[0];
      a1n = ((const uint4*)ap)[1];
    }
#endif

    // ---- compute current buffer
    Frag af[2], bfr[4];
#pragma unroll
    for (int mi = 0; mi < 2; ++mi) {
      const unsigned short* p = &sA[cur][(wm + mi * 16 + lrow) * 32 + kselA];
      af[mi].u[0] = ((const uint4*)p)[0];
      af[mi].u[1] = ((const uint4*)(p + 16))[0];
    }
#pragma unroll
    for (int ni = 0; ni < 4; ++ni) {
      const unsigned short* p = &sB[cur][(wn + ni * 16 + lrow) * 32 + kselB];
      bfr[ni].u[0] = ((const uint4*)p)[0];
      bfr[ni].u[1] = ((const uint4*)(p + 16))[0];
    }
#pragma unroll
    for (int mi = 0; mi < 2; ++mi)
#pragma unroll
      for (int ni = 0; ni < 4; ++ni)
        acc[mi][ni] = __builtin_amdgcn_wmma_f32_16x16x32_bf16(
            false, af[mi].v, false, bfr[ni].v, (short)0, acc[mi][ni], false, false);

    // ---- stage next buffer (no conflict with readers of `cur`)
    if (more) {
#ifndef HAVE_ASYNC_LDS
      ((uint4*)&sA[nxt][ar * 32 + ah])[0] = a0n;
      ((uint4*)&sA[nxt][ar * 32 + ah])[1] = a1n;
#endif
#pragma unroll
      for (int i = 0; i < 16; ++i) sB[nxt][(bn + i) * 32 + bk] = bw.s[i];
    }
    wait_async0();
    __syncthreads();
  }

  // ---- epilogue: swish (fast rcp), store bf16 H[slot][n]
  const int mhi = (lane >> 4) * 8;
#pragma unroll
  for (int mi = 0; mi < 2; ++mi) {
#pragma unroll
    for (int r = 0; r < 8; ++r) {
      const int m = wm + mi * 16 + mhi + r;
      const int slot = sSlot[m];                   // DUMMY row exists in Hb
      unsigned short* hrow = Hb + (size_t)slot * INTERF + nBase + wn;
#pragma unroll
      for (int ni = 0; ni < 4; ++ni) {
        union { v8f v; float f[8]; } cu; cu.v = acc[mi][ni];
        float v = cu.f[r];
        float sw = v * __builtin_amdgcn_rcpf(1.f + __expf(-v));
        hrow[ni * 16 + lrow] = f2bf(sw);
      }
    }
  }
}

// ---------------------------------------------------------------- GEMM2: Y[slot] = w * (H[slot] @ W2[e])
__global__ __launch_bounds__(256)
void gemm2_kernel(const unsigned short* __restrict__ Hb,
                  const unsigned short* __restrict__ W2b,
                  float* __restrict__ Y,
                  const int* __restrict__ list, const int* __restrict__ cnt,
                  const float* __restrict__ slot_w) {
  const int e = blockIdx.z;
  const int Ne = cnt[e];
  const int mBase = blockIdx.x * 128;
  if (mBase >= Ne) return;
  const int nBase = blockIdx.y * 128;              // over HID

  __shared__ unsigned short sA[2][128 * 32];
  __shared__ unsigned short sB[2][128 * 32];
  __shared__ int   sSlot[128];
  __shared__ float sW[128];

  const int t = threadIdx.x;
  if (t < 128) {
    int s = list[e * TOK + mBase + t];
    sSlot[t] = s;
    sW[t] = slot_w[s];                             // slot_w sized NSLOT+16, DUMMY safe
  }
  __syncthreads();

  const int ar = t >> 1, ah = (t & 1) * 16;
  const int bk = t >> 3, bn = (t & 7) * 16;
  const unsigned short* aSrc = Hb + (size_t)sSlot[ar] * INTERF + ah;
  const unsigned short* bSrc = W2b + (size_t)e * INTERF * HID + (size_t)bk * HID + nBase + bn;

  const int lane = t & 31;
  const int wid  = t >> 5;
  const int wm = (wid & 3) * 32;
  const int wn = (wid >> 2) * 64;
  const int lrow  = lane & 15;
  const int kselA = (lane < 16) ? 0 : 8;
  const int kselB = (lane < 16) ? 0 : 8;

  union Frag { uint4 u[2]; v16bf v; unsigned short s[16]; };
  v8f acc[2][4];
  const v8f vzero = {0.f,0.f,0.f,0.f,0.f,0.f,0.f,0.f};
#pragma unroll
  for (int mi = 0; mi < 2; ++mi)
#pragma unroll
    for (int ni = 0; ni < 4; ++ni) acc[mi][ni] = vzero;

#ifdef HAVE_ASYNC_LDS
  ld_async_b128(aSrc,     &sA[0][ar * 32 + ah]);
  ld_async_b128(aSrc + 8, &sA[0][ar * 32 + ah + 8]);
#else
  {
    uint4 a0 = ((const uint4*)aSrc)[0], a1 = ((const uint4*)aSrc)[1];
    ((uint4*)&sA[0][ar * 32 + ah])[0] = a0;
    ((uint4*)&sA[0][ar * 32 + ah])[1] = a1;
  }
#endif
  {
    Frag bw;
    bw.u[0] = ((const uint4*)bSrc)[0];
    bw.u[1] = ((const uint4*)bSrc)[1];
#pragma unroll
    for (int i = 0; i < 16; ++i) sB[0][(bn + i) * 32 + bk] = bw.s[i];
  }
  wait_async0();
  __syncthreads();

  const int KSTEPS = INTERF / 32;
  for (int s = 0; s < KSTEPS; ++s) {
    const int cur = s & 1, nxt = cur ^ 1;
    const bool more = (s + 1) < KSTEPS;

    Frag bw;
    if (more) {
      const unsigned short* bp = bSrc + (size_t)(s + 1) * 32 * HID;
      bw.u[0] = ((const uint4*)bp)[0];
      bw.u[1] = ((const uint4*)bp)[1];
#ifdef HAVE_ASYNC_LDS
      const unsigned short* ap = aSrc + (s + 1) * 32;
      ld_async_b128(ap,     &sA[nxt][ar * 32 + ah]);
      ld_async_b128(ap + 8, &sA[nxt][ar * 32 + ah + 8]);
#endif
    }
#ifndef HAVE_ASYNC_LDS
    uint4 a0n, a1n;
    if (more) {
      const unsigned short* ap = aSrc + (s + 1) * 32;
      a0n = ((const uint4*)ap)[0];
      a1n = ((const uint4*)ap)[1];
    }
#endif

    Frag af[2], bfr[4];
#pragma unroll
    for (int mi = 0; mi < 2; ++mi) {
      const unsigned short* p = &sA[cur][(wm + mi * 16 + lrow) * 32 + kselA];
      af[mi].u[0] = ((const uint4*)p)[0];
      af[mi].u[1] = ((const uint4*)(p + 16))[0];
    }
#pragma unroll
    for (int ni = 0; ni < 4; ++ni) {
      const unsigned short* p = &sB[cur][(wn + ni * 16 + lrow) * 32 + kselB];
      bfr[ni].u[0] = ((const uint4*)p)[0];
      bfr[ni].u[1] = ((const uint4*)(p + 16))[0];
    }
#pragma unroll
    for (int mi = 0; mi < 2; ++mi)
#pragma unroll
      for (int ni = 0; ni < 4; ++ni)
        acc[mi][ni] = __builtin_amdgcn_wmma_f32_16x16x32_bf16(
            false, af[mi].v, false, bfr[ni].v, (short)0, acc[mi][ni], false, false);

    if (more) {
#ifndef HAVE_ASYNC_LDS
      ((uint4*)&sA[nxt][ar * 32 + ah])[0] = a0n;
      ((uint4*)&sA[nxt][ar * 32 + ah])[1] = a1n;
#endif
#pragma unroll
      for (int i = 0; i < 16; ++i) sB[nxt][(bn + i) * 32 + bk] = bw.s[i];
    }
    wait_async0();
    __syncthreads();
  }

  // epilogue: weighted per-slot rows (each valid slot written exactly once -> deterministic)
  const int mhi = (lane >> 4) * 8;
#pragma unroll
  for (int mi = 0; mi < 2; ++mi) {
#pragma unroll
    for (int r = 0; r < 8; ++r) {
      const int m = wm + mi * 16 + mhi + r;
      const int slot = sSlot[m];
      if (slot < DUMMY) {
        const float w = sW[m];
        float* yrow = Y + (size_t)slot * HID + nBase + wn;
#pragma unroll
        for (int ni = 0; ni < 4; ++ni) {
          union { v8f v; float f[8]; } cu; cu.v = acc[mi][ni];
          yrow[ni * 16 + lrow] = w * cu.f[r];
        }
      }
    }
  }
}

// ---------------------------------------------------------------- combine the two expert outputs per token
__global__ void combine_kernel(const float* __restrict__ Y, float* __restrict__ out) {
  int i = blockIdx.x * blockDim.x + threadIdx.x;
  int stride = gridDim.x * blockDim.x;
  for (; i < TOK * HID; i += stride) {
    int tok = i >> 10, d = i & (HID - 1);
    out[i] = Y[(size_t)(tok * 2) * HID + d] + Y[(size_t)(tok * 2 + 1) * HID + d];
  }
}

// ---------------------------------------------------------------- aux scalars
__global__ void finalize_kernel(const int* __restrict__ cnt, const float* __restrict__ ent,
                                float* __restrict__ outTail) {
  const int lane = threadIdx.x;
  if (lane < NEXP) outTail[2 + lane] = (float)cnt[lane] / (float)(TOK * 2);
  float s = 0.f;
  for (int i = lane; i < TOK; i += 32) s += ent[i];
#pragma unroll
  for (int off = 16; off > 0; off >>= 1) s += __shfl_xor(s, off, 32);
  if (lane == 0) {
    outTail[1] = (s / (float)TOK) * 0.01f;         // entropy * ENTROPY_COEF
    float lb = 0.f;
    for (int e = 0; e < NEXP; ++e) {
      float u = (float)cnt[e] / (float)(TOK * 2) - 0.125f;
      lb += u * u;
    }
    outTail[0] = (lb / (float)NEXP) * 0.01f;       // load_balance * WEIGHT
  }
}

// ---------------------------------------------------------------- launch
extern "C" void kernel_launch(void* const* d_in, const int* in_sizes, int n_in,
                              void* d_out, int out_size, void* d_ws, size_t ws_size,
                              hipStream_t stream) {
  (void)in_sizes; (void)n_in; (void)out_size; (void)ws_size;
  const float* x  = (const float*)d_in[0];
  const float* Wg = (const float*)d_in[1];
  const float* W1 = (const float*)d_in[2];
  const float* W2 = (const float*)d_in[3];
  float* out = (float*)d_out;

  char* ws = (char*)d_ws;
  size_t off = 0;
  auto alloc = [&](size_t bytes) -> char* {
    char* p = ws + off;
    off += (bytes + 255) & ~(size_t)255;
    return p;
  };
  unsigned short* Xb  = (unsigned short*)alloc((size_t)TOK * HID * 2);
  unsigned short* W1b = (unsigned short*)alloc((size_t)NEXP * HID * INTERF * 2);
  unsigned short* W2b = (unsigned short*)alloc((size_t)NEXP * INTERF * HID * 2);
  unsigned short* Hb  = (unsigned short*)alloc((size_t)(NSLOT + 16) * INTERF * 2);
  float* Y     = (float*)alloc((size_t)(NSLOT + 16) * HID * 4);
  int*   list  = (int*)alloc((size_t)NEXP * TOK * 4);
  float* slotw = (float*)alloc((size_t)(NSLOT + 16) * 4);
  float* ent   = (float*)alloc((size_t)TOK * 4);
  int*   cnt   = (int*)alloc(256);

  init_kernel<<<1, 256, 0, stream>>>(list, cnt);
  cvt_bf16_kernel<<<2048, 256, 0, stream>>>(x,  Xb,  TOK * HID);
  cvt_bf16_kernel<<<4096, 256, 0, stream>>>(W1, W1b, NEXP * HID * INTERF);
  cvt_bf16_kernel<<<4096, 256, 0, stream>>>(W2, W2b, NEXP * INTERF * HID);
  router_kernel<<<TOK / 8, 256, 0, stream>>>(x, Wg, cnt, list, slotw, ent);

  dim3 g1(32, INTERF / 128, NEXP);
  gemm1_kernel<<<g1, 256, 0, stream>>>(Xb, W1b, Hb, list, cnt);
  dim3 g2(32, HID / 128, NEXP);
  gemm2_kernel<<<g2, 256, 0, stream>>>(Hb, W2b, Y, list, cnt, slotw);

  combine_kernel<<<2048, 256, 0, stream>>>(Y, out);
  finalize_kernel<<<1, 32, 0, stream>>>(cnt, ent, out + (size_t)TOK * HID);
}
